// FeatureMatching_10453950398628
// MI455X (gfx1250) — matrix operand order repaired and checked by reading, not compiled, not run on hardware
//
#include <hip/hip_runtime.h>
#include <hip/hip_bf16.h>
#include <math.h>

// ---------------------------------------------------------------------------
// FeatureMatching pipeline for gfx1250 (MI455X, wave32, WMMA), NHWC layout.
//  - Convs as implicit GEMM on v_wmma_f32_16x16x32_f16 (f16 ops, f32 accum).
//    NHWC (channels padded to 32) => each 32-wide K chunk = one conv tap with
//    contiguous channels: B-fragment is a single conditional 32B vector load,
//    output store is one B128 per lane.  Weight panels reordered to
//    k = tap*C + c at convert time and staged per-block into LDS via
//    global_load_async_to_lds_b128 (+ s_wait_asynccnt).
//  - Correlation (6400x144 x 144x6400 per batch) fused with max/argmax and
//    software-pipelined 2 deep (A-fragment double buffer) so key-tile loads
//    overlap WMMA; similarity matrix never materialized (~327MB avoided).
// ---------------------------------------------------------------------------

typedef __attribute__((ext_vector_type(16))) _Float16 v16h;
typedef __attribute__((ext_vector_type(8)))  _Float16 v8h;
typedef __attribute__((ext_vector_type(8)))  float    v8f;
typedef int v4i_vs __attribute__((vector_size(16)));   // matches builtin sig

union HFrag {
    v16h     v;
    v8h      q[2];
    _Float16 h[16];
    unsigned u[8];
};

#if defined(__HIP_DEVICE_COMPILE__) && __has_builtin(__builtin_amdgcn_global_load_async_to_lds_b128)
#define HAVE_ASYNC_LDS 1
#else
#define HAVE_ASYNC_LDS 0
#endif

#if HAVE_ASYNC_LDS
__device__ __forceinline__ void async_copy16(const _Float16* g, _Float16* l) {
    __builtin_amdgcn_global_load_async_to_lds_b128(
        (__attribute__((address_space(1))) v4i_vs*)(g),
        (__attribute__((address_space(3))) v4i_vs*)(l), 0, 0);
}
__device__ __forceinline__ void wait_async_zero() {
#if __has_builtin(__builtin_amdgcn_s_wait_asynccnt)
    __builtin_amdgcn_s_wait_asynccnt(0);
#else
    asm volatile("s_wait_asynccnt 0" ::: "memory");
#endif
}
#endif

#define A_COEF (-0.75f)

__device__ __forceinline__ float cubic_w(float d) {
    d = fabsf(d);
    float d2 = d * d;
    if (d <= 1.f) return ((A_COEF + 2.f) * d - (A_COEF + 3.f)) * d2 + 1.f;
    if (d <  2.f) return A_COEF * (((d - 5.f) * d + 8.f) * d - 4.f);
    return 0.f;
}

__constant__ float kMean[3]   = {0.485f, 0.456f, 0.406f};
__constant__ float kInvStd[3] = {1.f / 0.229f, 1.f / 0.224f, 1.f / 0.225f};

// ---------------------------------------------------------------------------
// Weight reorder+convert: f32 [O][CinReal][KK2] -> f16 [O][KK2][Cpad]
// (k = tap*Cpad + c; zero-fill padded channels).
// ---------------------------------------------------------------------------
__global__ void convert_weights_kernel(const float* __restrict__ W,
                                       _Float16* __restrict__ WH,
                                       int Cout, int CinReal, int Cpad, int KK2)
{
    int idx = blockIdx.x * blockDim.x + threadIdx.x;
    if (idx >= Cout * KK2 * Cpad) return;
    int c = idx % Cpad;
    int t = idx / Cpad;
    int j = t % KK2;
    int o = t / KK2;
    float v = (c < CinReal) ? W[((size_t)o * CinReal + c) * KK2 + j] : 0.f;
    WH[idx] = (_Float16)v;
}

// ---------------------------------------------------------------------------
// Preprocess: (optional avgpool2) -> sub_mean -> bicubic x2 (align_corners).
// Logical 160x160 -> 320x320, NHWC with C padded 3->32 (zeros), f16.
// One thread per output pixel; bicubic weights shared across channels.
// ---------------------------------------------------------------------------
__global__ void prep_kernel(const float* __restrict__ src,
                            _Float16* __restrict__ dst,
                            int B, int avgpool)
{
    const int OUT = 320, IN = 160, CP = 32;
    int idx = blockIdx.x * blockDim.x + threadIdx.x;
    if (idx >= B * OUT * OUT) return;

    int ox = idx % OUT;
    int t  = idx / OUT;
    int oy = t % OUT;
    int b  = t / OUT;

    const float scale = (float)(IN - 1) / (float)(OUT - 1);

    float cy = oy * scale;
    int   iy0 = (int)floorf(cy);
    float ty = cy - (float)iy0;
    float cx = ox * scale;
    int   ix0 = (int)floorf(cx);
    float tx = cx - (float)ix0;

    int   iy[4], ix[4];
    float wy[4], wx[4];
#pragma unroll
    for (int q = 0; q < 4; ++q) {
        int yy = iy0 - 1 + q;
        iy[q] = yy < 0 ? 0 : (yy > IN - 1 ? IN - 1 : yy);
        wy[q] = cubic_w(ty + 1.f - (float)q);
        int xx = ix0 - 1 + q;
        ix[q] = xx < 0 ? 0 : (xx > IN - 1 ? IN - 1 : xx);
        wx[q] = cubic_w(tx + 1.f - (float)q);
    }

    float accc[3] = {0.f, 0.f, 0.f};
#pragma unroll
    for (int qy = 0; qy < 4; ++qy) {
#pragma unroll
        for (int qx = 0; qx < 4; ++qx) {
            float w = wy[qy] * wx[qx];
#pragma unroll
            for (int c = 0; c < 3; ++c) {
                float v;
                if (avgpool) {
                    const float* p = src + (((size_t)(b * 3 + c) * 320 + iy[qy] * 2) * 320 + ix[qx] * 2);
                    v = 0.25f * (p[0] + p[1] + p[320] + p[321]);
                } else {
                    v = src[((size_t)(b * 3 + c) * IN + iy[qy]) * IN + ix[qx]];
                }
                accc[c] += w * (v - kMean[c]) * kInvStd[c];
            }
        }
    }

    HFrag f0 = {}, f1 = {};
    f0.h[0] = (_Float16)accc[0];
    f0.h[1] = (_Float16)accc[1];
    f0.h[2] = (_Float16)accc[2];
    _Float16* drow = dst + (size_t)idx * CP;
    *(v16h*)drow        = f0.v;
    *(v16h*)(drow + 16) = f1.v;
}

// ---------------------------------------------------------------------------
// Implicit-GEMM conv (stride 1), NHWC: Y[p, Cout] = im2col(X) * W^T.
// Block = 256 threads = 8 waves, tile = 16 out-channels x 128 pixels.
// 16 x Ktot f16 weight panel staged in LDS (async global->LDS), A-fragments
// from LDS as ds b128 pairs; B-fragment = one conditional 32B NHWC load.
// Pixels % 128 == 0, Cout % 16 == 0, Cin % 32 == 0 (padded).
// ---------------------------------------------------------------------------
template <int KS>
__global__ __launch_bounds__(256)
void conv_wmma_kernel(const _Float16* __restrict__ X,    // NHWC [B,H,W,Cin]
                      const _Float16* __restrict__ WH,   // f16 [Cout, KS*KS*Cin]
                      const float* __restrict__ Bias,
                      _Float16* __restrict__ Y,          // NHWC [B,H,W,Cout]
                      int Cin, int H, int W, int Cout, int leaky)
{
    constexpr int KK2 = KS * KS;
    extern __shared__ _Float16 smem[];

    const int Ktot    = Cin * KK2;                 // multiple of 32
    const int strideH = Ktot + 8;                  // 16B-aligned bank skew
    const int ch_base = (int)blockIdx.y * 16;

    // ---- stage 16 x Ktot weight panel into LDS ----------------------------
    {
        const int cpr = Ktot >> 3;                 // 16B chunks per row
        const int total_chunks = 16 * cpr;
        for (int t = (int)threadIdx.x; t < total_chunks; t += 256) {
            int row = t / cpr;
            int ko  = (t % cpr) << 3;
            const _Float16* src = WH + (size_t)(ch_base + row) * Ktot + ko;
            _Float16*       dst = smem + row * strideH + ko;
#if HAVE_ASYNC_LDS
            async_copy16(src, dst);
#else
            *(v8h*)dst = *(const v8h*)src;
#endif
        }
#if HAVE_ASYNC_LDS
        wait_async_zero();
#endif
        __syncthreads();
    }

    const int lane  = (int)(threadIdx.x & 31);
    const int wave  = (int)(threadIdx.x >> 5);
    const int lhalf = lane & 15;
    const int lhi   = lane >> 4;

    const int b  = (int)blockIdx.z;
    const int HW = H * W;
    const int p  = (int)blockIdx.x * 128 + wave * 16 + lhalf;  // pixel (col n)
    const int y  = p / W;
    const int x  = p % W;

    const _Float16* Xb   = X + (size_t)b * HW * Cin;
    const _Float16* arow = smem + lhalf * strideH;             // weight row m

    v8f acc = {0.f, 0.f, 0.f, 0.f, 0.f, 0.f, 0.f, 0.f};

    const int nCh = Ktot >> 5;
    for (int kc = 0; kc < nCh; ++kc) {
        const int kk   = kc << 5;
        const int kseg = kk + lhi * 16;
        int j0, c0;
        if (KS == 3) { j0 = kseg / Cin; c0 = kseg % Cin; }
        else         { j0 = 0;          c0 = kseg; }
        const int sy = y + (KS == 3 ? j0 / 3 - 1 : 0);
        const int sx = x + (KS == 3 ? j0 % 3 - 1 : 0);

        HFrag af, bf;
        // A from LDS: lanes 0-15 get K kk+0..15, lanes 16-31 kk+16..31
        af.q[0] = *(const v8h*)(arow + kk + lhi * 16);
        af.q[1] = *(const v8h*)(arow + kk + lhi * 16 + 8);
        // B: 16 contiguous NHWC channels at one conv tap (zero outside)
        v16h zv = {};
        const bool inb = (KS == 1) || (sy >= 0 && sy < H && sx >= 0 && sx < W);
        bf.v = inb ? *(const v16h*)(Xb + ((size_t)sy * W + sx) * Cin + c0) : zv;

        acc = __builtin_amdgcn_wmma_f32_16x16x32_f16(false, af.v, false, bf.v,
                                                     (short)0, acc, false, false);
    }

    // NHWC store: lane holds 8 contiguous channels for its pixel
    v8h ov;
#pragma unroll
    for (int r = 0; r < 8; ++r) {
        int   ch = ch_base + lhi * 8 + r;
        float v  = acc[r] + Bias[ch];
        v = leaky ? (v > 0.f ? v : 0.2f * v) : fmaxf(v, 0.f);
        ov[r] = (_Float16)v;
    }
    *(v8h*)(Y + ((size_t)b * HW + p) * Cout + ch_base + lhi * 8) = ov;
}

// ---------------------------------------------------------------------------
// 2x2 max pool, NHWC; one thread per 8-channel group => 4 vector loads.
// ---------------------------------------------------------------------------
__global__ void maxpool2_kernel(const _Float16* __restrict__ X,
                                _Float16* __restrict__ Y,
                                int B, int C, int Hi, int Wi)
{
    int Ho = Hi >> 1, Wo = Wi >> 1, CG = C >> 3;
    int idx = blockIdx.x * blockDim.x + threadIdx.x;
    if (idx >= B * Ho * Wo * CG) return;
    int cg = idx % CG;
    int t  = idx / CG;
    int x  = t % Wo; t /= Wo;
    int y  = t % Ho;
    int b  = t / Ho;
    const _Float16* base = X + (((size_t)b * Hi + 2 * y) * Wi + 2 * x) * C + cg * 8;
    v8h a0 = *(const v8h*)(base);
    v8h a1 = *(const v8h*)(base + C);
    v8h a2 = *(const v8h*)(base + (size_t)Wi * C);
    v8h a3 = *(const v8h*)(base + (size_t)Wi * C + C);
    v8h m;
#pragma unroll
    for (int r = 0; r < 8; ++r)
        m[r] = (_Float16)fmaxf(fmaxf((float)a0[r], (float)a1[r]),
                               fmaxf((float)a2[r], (float)a3[r]));
    *(v8h*)(Y + (((size_t)b * Ho + y) * Wo + x) * C + cg * 8) = m;
}

// ---------------------------------------------------------------------------
// Unfold 3x3 (reflect pad) + L2 normalize, NHWC C=16 input.
// K ordering: k = tap*16 + c (consistent on both paths => same dot products).
// Output rows: [b][l][160] halves (K padded 144->160 with zeros, 32B aligned).
// ---------------------------------------------------------------------------
__global__ void patches_norm_kernel(const _Float16* __restrict__ F,  // NHWC
                                    _Float16* __restrict__ P, int B)
{
    int idx = blockIdx.x * blockDim.x + threadIdx.x;
    if (idx >= B * 6400) return;
    int b = idx / 6400, l = idx % 6400;
    int y = l / 80, x = l % 80;
    const _Float16* Fb = F + (size_t)b * 6400 * 16;

    float acc = 0.f;
#pragma unroll
    for (int jj = 0; jj < 9; ++jj) {
        int sy = y + jj / 3 - 1; sy = sy < 0 ? -sy : (sy > 79 ? 158 - sy : sy);
        int sx = x + jj % 3 - 1; sx = sx < 0 ? -sx : (sx > 79 ? 158 - sx : sx);
        const _Float16* src = Fb + ((size_t)sy * 80 + sx) * 16;
#pragma unroll
        for (int c = 0; c < 16; ++c) {
            float v = (float)src[c];
            acc += v * v;
        }
    }
    float inv = 1.f / fmaxf(sqrtf(acc), 1e-12f);

    _Float16* row = P + ((size_t)b * 6400 + l) * 160;
#pragma unroll
    for (int jj = 0; jj < 9; ++jj) {
        int sy = y + jj / 3 - 1; sy = sy < 0 ? -sy : (sy > 79 ? 158 - sy : sy);
        int sx = x + jj % 3 - 1; sx = sx < 0 ? -sx : (sx > 79 ? 158 - sx : sx);
        const _Float16* src = Fb + ((size_t)sy * 80 + sx) * 16;
        HFrag w;
#pragma unroll
        for (int c = 0; c < 16; ++c)
            w.h[c] = (_Float16)((float)src[c] * inv);
        *(v16h*)(row + jj * 16) = w.v;
    }
    HFrag z = {};
    *(v16h*)(row + 144) = z.v;
}

// ---------------------------------------------------------------------------
// Fused correlation GEMM + max/argmax, software-pipelined 2 deep.
// Each wave: 16 query columns (B-fragments hoisted, 5 x v16h); 400 key tiles
// streamed through a double-buffered A-fragment set so tile mt+1's loads
// overlap tile mt's WMMAs.  C layout reduced with a lane^16 shuffle (wave32).
// ---------------------------------------------------------------------------
__device__ __forceinline__ void load_key_tile(const _Float16* KPb, int mt,
                                              int lhalf, int lhi, HFrag* dst)
{
    const _Float16* krow = KPb + (size_t)(mt * 16 + lhalf) * 160;
    if (mt < 398)
        __builtin_prefetch(krow + 2 * 16 * 160, 0, 3);
#pragma unroll
    for (int kg = 0; kg < 5; ++kg) {
        dst[kg].q[0] = *(const v8h*)(krow + kg * 32 + lhi * 16);
        dst[kg].q[1] = *(const v8h*)(krow + kg * 32 + lhi * 16 + 8);
    }
}

__global__ __launch_bounds__(256)
void corr_argmax_kernel(const _Float16* __restrict__ KP,
                        const _Float16* __restrict__ QN,
                        float* __restrict__ rel,
                        float* __restrict__ idxOut)
{
    const int lane  = (int)(threadIdx.x & 31);
    const int wave  = (int)(threadIdx.x >> 5);
    const int lhalf = lane & 15;
    const int lhi   = lane >> 4;
    const int b     = (int)blockIdx.y;

    const int lq = (int)blockIdx.x * 128 + wave * 16 + lhalf;
    const _Float16* qrow = QN + ((size_t)b * 6400 + lq) * 160;

    v16h bf[5];
#pragma unroll
    for (int kg = 0; kg < 5; ++kg)
        bf[kg] = *(const v16h*)(qrow + kg * 32 + lhi * 16);

    const _Float16* KPb = KP + (size_t)b * 6400 * 160;

    float best = -3.0e38f;
    int bestIdx = 0;

    HFrag abuf[2][5];
    load_key_tile(KPb, 0, lhalf, lhi, abuf[0]);

    for (int mt2 = 0; mt2 < 400; mt2 += 2) {
#pragma unroll
        for (int par = 0; par < 2; ++par) {
            const int mt = mt2 + par;
            if (mt + 1 < 400)
                load_key_tile(KPb, mt + 1, lhalf, lhi, abuf[par ^ 1]);

            v8f acc = {0.f, 0.f, 0.f, 0.f, 0.f, 0.f, 0.f, 0.f};
#pragma unroll
            for (int kg = 0; kg < 5; ++kg)
                acc = __builtin_amdgcn_wmma_f32_16x16x32_f16(false, abuf[par][kg].v,
                                                             false, bf[kg],
                                                             (short)0, acc, false, false);

            // lane-local reduce (m = r + 8*lhi), first-max-wins tie-break
            float bv = -3.0e38f; int bm = 0;
#pragma unroll
            for (int r = 0; r < 8; ++r) {
                float v = acc[r];
                if (v > bv) { bv = v; bm = r + lhi * 8; }
            }
            // combine the two lanes holding the same column n (lane ^ 16)
            float ov = __shfl_xor(bv, 16, 32);
            int   om = __shfl_xor(bm, 16, 32);
            if (ov > bv || (ov == bv && om < bm)) { bv = ov; bm = om; }

            if (bv > best) { best = bv; bestIdx = mt * 16 + bm; }
        }
    }

    if (lhi == 0) {
        rel[(size_t)b * 6400 + lq]    = best;
        idxOut[(size_t)b * 6400 + lq] = (float)bestIdx;
    }
}

// ---------------------------------------------------------------------------
// Host-side orchestration.  Workspace layout (256B aligned offsets):
//   wbuf:  f16 reordered weights        (  557,056 B)
//   buf0:  NHWC 2x320x320x32 f16        (13,107,200 B)
//   bufA:  NHWC ping (max 2x320x320x64) (26,214,400 B)
//   bufB:  NHWC pong                    (26,214,400 B)
//   qbuf:  2*6400*160 f16               ( 4,096,000 B)
//   kbuf:  2*6400*160 f16               ( 4,096,000 B)
//   total ~71 MB
// ---------------------------------------------------------------------------
extern "C" void kernel_launch(void* const* d_in, const int* in_sizes, int n_in,
                              void* d_out, int out_size, void* d_ws, size_t ws_size,
                              hipStream_t stream)
{
    (void)in_sizes; (void)n_in; (void)out_size; (void)ws_size;

    const float* query = (const float*)d_in[0];
    const float* key   = (const float*)d_in[1];
    const float* w1 = (const float*)d_in[2];  const float* b1 = (const float*)d_in[3];
    const float* w2 = (const float*)d_in[4];  const float* b2 = (const float*)d_in[5];
    const float* w3 = (const float*)d_in[6];  const float* b3 = (const float*)d_in[7];
    const float* w4 = (const float*)d_in[8];  const float* b4 = (const float*)d_in[9];
    const float* wm = (const float*)d_in[10]; const float* bm = (const float*)d_in[11];

    const int B = 2;
    char* ws = (char*)d_ws;
    _Float16* wbuf = (_Float16*)(ws);                  //    557,056 B
    _Float16* buf0 = (_Float16*)(ws + 557056);         // 13,107,200 B
    _Float16* bufA = (_Float16*)(ws + 13664256);       // 26,214,400 B
    _Float16* bufB = (_Float16*)(ws + 39878656);       // 26,214,400 B
    _Float16* qbuf = (_Float16*)(ws + 66093056);       //  4,096,000 B
    _Float16* kbuf = (_Float16*)(ws + 70189056);       //  4,096,000 B

    // reordered f16 weight panels inside wbuf (k = tap*Cpad + c)
    _Float16* wh1 = wbuf;                  // 64  x 9*32  = 18432
    _Float16* wh2 = wh1 + 64 * 288;        // 64  x 9*64  = 36864
    _Float16* wh3 = wh2 + 64 * 576;        // 128 x 9*64  = 73728
    _Float16* wh4 = wh3 + 128 * 576;       // 128 x 9*128 = 147456
    _Float16* whm = wh4 + 128 * 1152;      // 16  x 128   = 2048

    auto cvt = [&](const float* w, _Float16* wh, int Cout, int CinReal, int Cpad, int KK2) {
        int total = Cout * KK2 * Cpad;
        convert_weights_kernel<<<(total + 255) / 256, 256, 0, stream>>>(
            w, wh, Cout, CinReal, Cpad, KK2);
    };
    cvt(w1, wh1, 64, 3, 32, 9);
    cvt(w2, wh2, 64, 64, 64, 9);
    cvt(w3, wh3, 128, 64, 64, 9);
    cvt(w4, wh4, 128, 128, 128, 9);
    cvt(wm, whm, 16, 128, 128, 1);

    auto smem_bytes = [](int Ktot) {
        return (size_t)16 * (Ktot + 8) * sizeof(_Float16);
    };

    auto run_path = [&](const float* src, int avg, _Float16* pbuf) {
        {
            int total = B * 320 * 320;
            prep_kernel<<<(total + 255) / 256, 256, 0, stream>>>(src, buf0, B, avg);
        }
        // conv1: 32(pad3)->64 @320
        conv_wmma_kernel<3><<<dim3(102400 / 128, 4, B), 256, smem_bytes(288), stream>>>(
            buf0, wh1, b1, bufA, 32, 320, 320, 64, 0);
        // conv2: 64->64 @320
        conv_wmma_kernel<3><<<dim3(102400 / 128, 4, B), 256, smem_bytes(576), stream>>>(
            bufA, wh2, b2, bufB, 64, 320, 320, 64, 0);
        // maxpool 320->160 (NHWC, 8ch per thread)
        {
            int total = B * 160 * 160 * (64 / 8);
            maxpool2_kernel<<<(total + 255) / 256, 256, 0, stream>>>(bufB, bufA, B, 64, 320, 320);
        }
        // conv3: 64->128 @160
        conv_wmma_kernel<3><<<dim3(25600 / 128, 8, B), 256, smem_bytes(576), stream>>>(
            bufA, wh3, b3, bufB, 64, 160, 160, 128, 0);
        // conv4: 128->128 @160
        conv_wmma_kernel<3><<<dim3(25600 / 128, 8, B), 256, smem_bytes(1152), stream>>>(
            bufB, wh4, b4, bufA, 128, 160, 160, 128, 0);
        // maxpool 160->80
        {
            int total = B * 80 * 80 * (128 / 8);
            maxpool2_kernel<<<(total + 255) / 256, 256, 0, stream>>>(bufA, bufB, B, 128, 160, 160);
        }
        // 1x1 map conv 128->16 + LeakyReLU(0.2)
        conv_wmma_kernel<1><<<dim3(6400 / 128, 1, B), 256, smem_bytes(128), stream>>>(
            bufB, whm, bm, bufA, 128, 80, 80, 16, 1);
        // unfold 3x3 reflect + L2 normalize -> pbuf [b][6400][160]
        {
            int total = B * 6400;
            patches_norm_kernel<<<(total + 255) / 256, 256, 0, stream>>>(bufA, pbuf, B);
        }
    };

    run_path(query, /*avg=*/0, qbuf);
    run_path(key,   /*avg=*/1, kbuf);

    float* rel = (float*)d_out;           // (B,1,80,80) flat = B*6400
    float* idx = rel + (size_t)B * 6400;  // hard indices (written as values)
    corr_argmax_kernel<<<dim3(6400 / 128, B), 256, 0, stream>>>(kbuf, qbuf, rel, idx);
}
